// PairSAE_35622458753220
// MI455X (gfx1250) — matrix operand order repaired and checked by reading, not compiled, not run on hardware
//
#include <hip/hip_runtime.h>
#include <hip/hip_bf16.h>

// ---------------------------------------------------------------------------
// Sizes (match reference)
// ---------------------------------------------------------------------------
#define D_IN   4096
#define WIDTH  32768
#define BATCH  4096
#define TOPK   64

// LDS staging geometry (bf16 elements)
#define TILE_ROWS   128
#define TILE_K      32
#define ROW_STRIDE  40                    // 32 data + 4-DWORD TDM pad -> 80 B, bank-conflict-free
#define TENSOR_ELEMS (TILE_ROWS * ROW_STRIDE)       // 5120
#define STAGE_ELEMS  (4 * TENSOR_ELEMS)             // 20480  (xh, xl, wh, wl)
#define NSTAGE       3
#define SMEM_BYTES   (NSTAGE * STAGE_ELEMS * 2)     // 122880 B

// ---------------------------------------------------------------------------
// Vector types
// ---------------------------------------------------------------------------
typedef __attribute__((ext_vector_type(16))) __bf16 bf16x16;
typedef __attribute__((ext_vector_type(8)))  __bf16 bf16x8;
typedef __attribute__((ext_vector_type(8)))  float  floatx8;
typedef __attribute__((ext_vector_type(4)))  unsigned int uint32x4;
typedef __attribute__((ext_vector_type(8)))  int    int32x8;
typedef __attribute__((ext_vector_type(4)))  int    int32x4;

// ---------------------------------------------------------------------------
// Kernel 1: split fp32 -> (hi, lo) bf16 pair.  v = hi + lo to ~fp32 accuracy.
// ---------------------------------------------------------------------------
__global__ __launch_bounds__(256)
void split_bf16_kernel(const float* __restrict__ in,
                       unsigned short* __restrict__ hi_u,
                       unsigned short* __restrict__ lo_u,
                       long long n)
{
    long long i = ((long long)blockIdx.x * 256 + threadIdx.x) * 8;
    if (i >= n) return;
    __bf16* hi = reinterpret_cast<__bf16*>(hi_u);
    __bf16* lo = reinterpret_cast<__bf16*>(lo_u);

    float4 a = *(const float4*)(in + i);
    float4 b = *(const float4*)(in + i + 4);
    float v[8] = {a.x, a.y, a.z, a.w, b.x, b.y, b.z, b.w};
    bf16x8 h, l;
#pragma unroll
    for (int e = 0; e < 8; ++e) {
        __bf16 hb = (__bf16)v[e];
        h[e] = hb;
        l[e] = (__bf16)(v[e] - (float)hb);
    }
    *(bf16x8*)(hi + i) = h;
    *(bf16x8*)(lo + i) = l;
}

// ---------------------------------------------------------------------------
// TDM descriptor issue: 2-D bf16 tensor, tile = TILE_K x TILE_ROWS, with LDS
// row padding 64B data + 16B pad (interval code 3 = 16 DWORDs, amount code 3
// = 4 DWORDs) so LDS row stride is 80 B. (ISA 8.3-8.5 field packing.)
// This toolchain exposes the 6-arg builtin (clang-23 / therock-10.0 form):
//   (uint32x4 g0, int32x8 g1, int32x4 g2, int32x4 g3, int32x8 gx, i32 cpol)
// Groups 2/3 (dims 2..4) and the extra group are zero for a 2-D tensor.
// ---------------------------------------------------------------------------
__device__ __forceinline__
void tdm_load_tile(unsigned lds_byte_off, const unsigned short* gptr,
                   unsigned tensor_rows)
{
    unsigned long long ga = (unsigned long long)(uintptr_t)gptr;
    uint32x4 g0;
    g0[0] = 1u;                                           // count=1, user mode
    g0[1] = lds_byte_off;                                 // lds_addr
    g0[2] = (unsigned)(ga & 0xFFFFFFFFull);               // global_addr[31:0]
    g0[3] = (unsigned)((ga >> 32) & 0x01FFFFFFull)        // global_addr[56:32]
          | 0x80000000u;                                  // type=2 ("image")

    int32x8 g1;
    // [17:16]=data_size(1=2B) [20]=pad_en [24:22]=pad_interval=3 [31:25]=pad_amount=3
    g1[0] = (int)0x06D10000u;
    g1[1] = (int)((D_IN & 0xFFFFu) << 16);                // tensor_dim0 lo16
    g1[2] = (int)(((D_IN >> 16) & 0xFFFFu)                // tensor_dim0 hi16
          | ((tensor_rows & 0xFFFFu) << 16));             // tensor_dim1 lo16
    g1[3] = (int)(((tensor_rows >> 16) & 0xFFFFu)         // tensor_dim1 hi16
          | ((unsigned)TILE_K << 16));                    // tile_dim0 = 32
    g1[4] = (int)TILE_ROWS;                               // tile_dim1 = 128, tile_dim2 = 0
    g1[5] = (int)D_IN;                                    // tensor_dim0_stride lo32
    g1[6] = 0;
    g1[7] = 0;

    int32x4 gz4 = {0, 0, 0, 0};                           // dims 2..4 unused (2-D)
    int32x8 gz8 = {0, 0, 0, 0, 0, 0, 0, 0};
#if __has_builtin(__builtin_amdgcn_tensor_load_to_lds)
    __builtin_amdgcn_tensor_load_to_lds(g0, g1, gz4, gz4, gz8, 0);
#endif
}

// ---------------------------------------------------------------------------
// Kernel 2: encode GEMM  z = relu(x @ W^T + b_enc) via bf16x3 WMMA, with
// TDM async staging of operand tiles into triple-buffered LDS.
//   Workgroup tile 128(M) x 128(N); 8 waves (4Mx2N), wave tile 32x64.
//   Waves 0..3 each issue one tensor_load_to_lds per stage (xh/xl/wh/wl).
// ---------------------------------------------------------------------------
__global__ __launch_bounds__(256)
void encode_wmma_kernel(const unsigned short* __restrict__ xh_u,
                        const unsigned short* __restrict__ xl_u,
                        const unsigned short* __restrict__ wh_u,
                        const unsigned short* __restrict__ wl_u,
                        const float* __restrict__ b_enc,
                        float* __restrict__ z)
{
    extern __shared__ __bf16 smem[];      // NSTAGE * STAGE_ELEMS bf16

    const int lane  = threadIdx.x & 31;
    const int wv    = threadIdx.x >> 5;
    const int waveM = wv >> 1;            // 0..3
    const int waveN = wv & 1;             // 0..1
    const int g     = lane >> 4;          // lane half
    const int l16   = lane & 15;

    const int m0 = blockIdx.x * 128;      // over BATCH
    const int n0 = blockIdx.y * 128;      // over WIDTH

    const unsigned lds_base = (unsigned)(uintptr_t)&smem[0];

    // per-wave TDM source (waves 0..3 only): tensor base row for this block
    const unsigned short* gsrc = nullptr;
    unsigned grows = 0;
    unsigned t_off = 0;                   // tensor slot byte offset within a stage
    if (wv == 0)      { gsrc = xh_u + (size_t)m0 * D_IN; grows = BATCH; t_off = 0 * TENSOR_ELEMS * 2; }
    else if (wv == 1) { gsrc = xl_u + (size_t)m0 * D_IN; grows = BATCH; t_off = 1 * TENSOR_ELEMS * 2; }
    else if (wv == 2) { gsrc = wh_u + (size_t)n0 * D_IN; grows = WIDTH; t_off = 2 * TENSOR_ELEMS * 2; }
    else if (wv == 3) { gsrc = wl_u + (size_t)n0 * D_IN; grows = WIDTH; t_off = 3 * TENSOR_ELEMS * 2; }

    floatx8 acc[2][4];
#pragma unroll
    for (int mt = 0; mt < 2; ++mt)
#pragma unroll
        for (int nt = 0; nt < 4; ++nt)
            acc[mt][nt] = (floatx8)0.0f;

    const int NSTEP = D_IN / TILE_K;      // 128

    // ---- prologue: put two stages in flight ----
    if (wv < 4) {
        tdm_load_tile(lds_base + 0 * STAGE_ELEMS * 2 + t_off, gsrc + 0 * TILE_K, grows);
        tdm_load_tile(lds_base + 1 * STAGE_ELEMS * 2 + t_off, gsrc + 1 * TILE_K, grows);
    }

    for (int s = 0; s < NSTEP; ++s) {
        const int buf = s % NSTAGE;
        // oldest outstanding tensor load (stage s) must be complete
        if (wv < 4) {
            if (s + 2 < NSTEP) __builtin_amdgcn_s_wait_tensorcnt((short)1);
            else               __builtin_amdgcn_s_wait_tensorcnt((short)0);
        }
        __syncthreads();                  // stage s visible to all waves

        // issue stage s+2 into buffer consumed at iteration s-1
        if (wv < 4 && s + 2 < NSTEP) {
            const int nbuf = (s + 2) % NSTAGE;
            tdm_load_tile(lds_base + nbuf * STAGE_ELEMS * 2 + t_off,
                          gsrc + (size_t)(s + 2) * TILE_K, grows);
        }

        const int sb = buf * STAGE_ELEMS;

        // ---- A fragments (hi & lo) for 2 M-tiles, from LDS ----
        bf16x16 Afrag[2][2];
#pragma unroll
        for (int mt = 0; mt < 2; ++mt) {
            const int row = waveM * 32 + mt * 16 + l16;
            const int bh  = sb + 0 * TENSOR_ELEMS + row * ROW_STRIDE + 16 * g;
            const int bl  = sb + 1 * TENSOR_ELEMS + row * ROW_STRIDE + 16 * g;
            bf16x8 h0 = *(const bf16x8*)(smem + bh);
            bf16x8 h1 = *(const bf16x8*)(smem + bh + 8);
            bf16x8 l0 = *(const bf16x8*)(smem + bl);
            bf16x8 l1 = *(const bf16x8*)(smem + bl + 8);
            Afrag[mt][0] = __builtin_shufflevector(h0, h1, 0,1,2,3,4,5,6,7,8,9,10,11,12,13,14,15);
            Afrag[mt][1] = __builtin_shufflevector(l0, l1, 0,1,2,3,4,5,6,7,8,9,10,11,12,13,14,15);
        }
        // ---- B fragments (hi & lo) for 4 N-tiles, from LDS ----
        bf16x16 Bfrag[4][2];
#pragma unroll
        for (int nt = 0; nt < 4; ++nt) {
            const int row = waveN * 64 + nt * 16 + l16;
            const int bh  = sb + 2 * TENSOR_ELEMS + row * ROW_STRIDE + 16 * g;
            const int bl  = sb + 3 * TENSOR_ELEMS + row * ROW_STRIDE + 16 * g;
            bf16x8 h0 = *(const bf16x8*)(smem + bh);
            bf16x8 h1 = *(const bf16x8*)(smem + bh + 8);
            bf16x8 l0 = *(const bf16x8*)(smem + bl);
            bf16x8 l1 = *(const bf16x8*)(smem + bl + 8);
            Bfrag[nt][0] = __builtin_shufflevector(h0, h1, 0,1,2,3,4,5,6,7,8,9,10,11,12,13,14,15);
            Bfrag[nt][1] = __builtin_shufflevector(l0, l1, 0,1,2,3,4,5,6,7,8,9,10,11,12,13,14,15);
        }
        // ---- 24 WMMAs: hi*hi + hi*lo + lo*hi ----
#pragma unroll
        for (int mt = 0; mt < 2; ++mt) {
#pragma unroll
            for (int nt = 0; nt < 4; ++nt) {
                acc[mt][nt] = __builtin_amdgcn_wmma_f32_16x16x32_bf16(
                    false, Afrag[mt][0], false, Bfrag[nt][0], (short)0, acc[mt][nt], false, false);
                acc[mt][nt] = __builtin_amdgcn_wmma_f32_16x16x32_bf16(
                    false, Afrag[mt][0], false, Bfrag[nt][1], (short)0, acc[mt][nt], false, false);
                acc[mt][nt] = __builtin_amdgcn_wmma_f32_16x16x32_bf16(
                    false, Afrag[mt][1], false, Bfrag[nt][0], (short)0, acc[mt][nt], false, false);
            }
        }
        __syncthreads();                  // all reads of buf done before reuse
    }

    // ---- store with bias + ReLU. C layout: VGPR v -> M = v + 8*g, N = l16 ----
#pragma unroll
    for (int mt = 0; mt < 2; ++mt) {
#pragma unroll
        for (int nt = 0; nt < 4; ++nt) {
            const int col  = n0 + waveN * 64 + nt * 16 + l16;
            const float bias = b_enc[col];
#pragma unroll
            for (int v = 0; v < 8; ++v) {
                const int rowm = m0 + waveM * 32 + mt * 16 + 8 * g + v;
                float val = acc[mt][nt][v] + bias;
                z[(size_t)rowm * WIDTH + col] = val > 0.0f ? val : 0.0f;
            }
        }
    }
}

// ---------------------------------------------------------------------------
// Kernel 3: per-row top-64 radix select (post-ReLU values >= 0, so float
// ordering == uint ordering). Zeroes non-kept entries of z in place, emits a
// compacted (idx,val) list in deterministic index-ascending order; ties at
// the threshold broken by lowest index (matches lax.top_k + scatter).
// ---------------------------------------------------------------------------
__global__ __launch_bounds__(256)
void topk64_kernel(float* __restrict__ z,
                   int* __restrict__ out_idx,
                   float* __restrict__ out_val)
{
    const int row = blockIdx.x;
    const int tid = threadIdx.x;
    float* zr = z + (size_t)row * WIDTH;

    __shared__ int hist[256];
    __shared__ int sh_g[256], sh_e[256], sh_eoff[256], sh_koff[256];
    __shared__ unsigned sh_prefix, sh_mask;
    __shared__ int sh_want, sh_eqkeep;

    if (tid == 0) { sh_prefix = 0u; sh_mask = 0u; sh_want = TOPK; }
    __syncthreads();

    for (int shift = 24; shift >= 0; shift -= 8) {
        hist[tid] = 0;
        __syncthreads();
        const unsigned prefix = sh_prefix;
        const unsigned mask   = sh_mask;
        for (int i = tid; i < WIDTH; i += 256) {
            unsigned key = __float_as_uint(zr[i]);
            if ((key & mask) == prefix)
                atomicAdd(&hist[(key >> shift) & 255], 1);
        }
        __syncthreads();
        if (tid == 0) {
            int want = sh_want, cum = 0, sel = 0;
            for (int b = 255; b >= 0; --b) {
                int c = hist[b];
                if (cum + c >= want) { sel = b; sh_want = want - cum; break; }
                cum += c;
            }
            sh_prefix = prefix | ((unsigned)sel << shift);
            sh_mask   = mask | (255u << shift);
        }
        __syncthreads();
    }
    const unsigned thresh = sh_prefix;

    const int chunk = WIDTH / 256;        // 128
    const int i0 = tid * chunk;
    int g_t = 0, e_t = 0;
    for (int i = i0; i < i0 + chunk; ++i) {
        unsigned key = __float_as_uint(zr[i]);
        if (key > thresh) ++g_t;
        else if (key == thresh) ++e_t;
    }
    sh_g[tid] = g_t;
    sh_e[tid] = e_t;
    __syncthreads();

    if (tid == 0) {
        int ag = 0, ae = 0;
        for (int t = 0; t < 256; ++t) {
            sh_eoff[t] = ae;
            ae += sh_e[t];
            ag += sh_g[t];
        }
        int eqk = TOPK - ag;
        if (eqk < 0) eqk = 0;
        sh_eqkeep = eqk;
        int ak = 0;
        for (int t = 0; t < 256; ++t) {
            int avail = eqk - sh_eoff[t];
            if (avail < 0) avail = 0;
            if (avail > sh_e[t]) avail = sh_e[t];
            sh_koff[t] = ak;
            ak += sh_g[t] + avail;
        }
    }
    __syncthreads();

    const int eq_keep = sh_eqkeep;
    int pos   = sh_koff[tid];
    int erank = sh_eoff[tid];
    for (int i = i0; i < i0 + chunk; ++i) {
        float v = zr[i];
        unsigned key = __float_as_uint(v);
        bool keep = (key > thresh);
        if (!keep && key == thresh) {
            keep = (erank < eq_keep);
            ++erank;
        }
        if (keep) {
            out_idx[row * TOPK + pos] = i;
            out_val[row * TOPK + pos] = v;
            ++pos;
        } else {
            zr[i] = 0.0f;
        }
    }
}

// ---------------------------------------------------------------------------
// Kernel 4: sparse decode  x_hat[b,:] = b_dec + sum_j val[b,j] * W[idx[b,j],:]
// ---------------------------------------------------------------------------
__global__ __launch_bounds__(256)
void decode64_kernel(const float* __restrict__ W,
                     const float* __restrict__ b_dec,
                     const int* __restrict__ idx,
                     const float* __restrict__ val,
                     float* __restrict__ xhat)
{
    const int row = blockIdx.x;
    const int tid = threadIdx.x;
    __shared__ int   s_idx[TOPK];
    __shared__ float s_val[TOPK];
    if (tid < TOPK) {
        s_idx[tid] = idx[row * TOPK + tid];
        s_val[tid] = val[row * TOPK + tid];
    }
    __syncthreads();

    float acc[16];
#pragma unroll
    for (int u = 0; u < 16; ++u) acc[u] = b_dec[tid + 256 * u];

    for (int j = 0; j < TOPK; ++j) {
        const float* wr = W + (size_t)s_idx[j] * D_IN;
        const float  sv = s_val[j];
#pragma unroll
        for (int u = 0; u < 16; ++u)
            acc[u] = fmaf(sv, wr[tid + 256 * u], acc[u]);
    }
#pragma unroll
    for (int u = 0; u < 16; ++u)
        xhat[(size_t)row * D_IN + tid + 256 * u] = acc[u];
}

// ---------------------------------------------------------------------------
// Launch
// ---------------------------------------------------------------------------
extern "C" void kernel_launch(void* const* d_in, const int* in_sizes, int n_in,
                              void* d_out, int out_size, void* d_ws, size_t ws_size,
                              hipStream_t stream)
{
    const float* x     = (const float*)d_in[0];   // [BATCH, D_IN]
    const float* W     = (const float*)d_in[1];   // [WIDTH, D_IN]
    const float* b_enc = (const float*)d_in[2];   // [WIDTH]
    const float* b_dec = (const float*)d_in[3];   // [D_IN]
    // d_in[4] = topk scalar (=64), hardcoded as TOPK

    float* z    = (float*)d_out;                  // [BATCH, WIDTH]
    float* xhat = z + (size_t)BATCH * WIDTH;      // [BATCH, D_IN]

    char* ws = (char*)d_ws;
    unsigned short* xh = (unsigned short*)ws;  ws += (size_t)BATCH * D_IN * 2;
    unsigned short* xl = (unsigned short*)ws;  ws += (size_t)BATCH * D_IN * 2;
    unsigned short* wh = (unsigned short*)ws;  ws += (size_t)WIDTH * D_IN * 2;
    unsigned short* wl = (unsigned short*)ws;  ws += (size_t)WIDTH * D_IN * 2;
    int*   idx = (int*)ws;                     ws += (size_t)BATCH * TOPK * 4;
    float* val = (float*)ws;

    // 1) fp32 -> bf16 hi/lo split
    {
        long long nx = (long long)BATCH * D_IN;   // 16.7M
        long long nw = (long long)WIDTH * D_IN;   // 134M
        split_bf16_kernel<<<(unsigned)(nx / (8 * 256)), 256, 0, stream>>>(x, xh, xl, nx);
        split_bf16_kernel<<<(unsigned)(nw / (8 * 256)), 256, 0, stream>>>(W, wh, wl, nw);
    }

    // 2) encode GEMM (TDM-staged, WMMA bf16x3) + bias + ReLU
    {
        dim3 grid(BATCH / 128, WIDTH / 128);      // 32 x 256
        encode_wmma_kernel<<<grid, 256, SMEM_BYTES, stream>>>(xh, xl, wh, wl, b_enc, z);
    }

    // 3) top-64 per row
    topk64_kernel<<<BATCH, 256, 0, stream>>>(z, idx, val);

    // 4) sparse decode
    decode64_kernel<<<BATCH, 256, 0, stream>>>(W, b_dec, idx, val, xhat);
}
